// GAT_27212912788335
// MI455X (gfx1250) — matrix operand order repaired and checked by reading, not compiled, not run on hardware
//
#include <hip/hip_runtime.h>
#include <hip/hip_bf16.h>

typedef __attribute__((ext_vector_type(16))) __bf16 v16bf;
typedef __attribute__((ext_vector_type(8)))  float  v8f;

#define NEG_SLOPE 0.2f
#define GAT_EPS 1e-16f

__device__ __forceinline__ unsigned short f2bf(float f) {
    unsigned u = __float_as_uint(f);
    unsigned r = u + 0x7fffu + ((u >> 16) & 1u);   // round-to-nearest-even
    return (unsigned short)(r >> 16);
}
// order-preserving float -> uint key for atomicMax
__device__ __forceinline__ unsigned fkey(float f) {
    unsigned u = __float_as_uint(f);
    return (u & 0x80000000u) ? ~u : (u | 0x80000000u);
}
__device__ __forceinline__ float funkey(unsigned k) {
    unsigned u = (k & 0x80000000u) ? (k ^ 0x80000000u) : ~k;
    return __uint_as_float(u);
}
__device__ __forceinline__ float lrelu(float v) { return v > 0.0f ? v : NEG_SLOPE * v; }

__global__ void zero_kernel(float* p, long long n) {
    long long i = (long long)blockIdx.x * blockDim.x + threadIdx.x;
    long long stride = (long long)gridDim.x * blockDim.x;
    for (; i < n; i += stride) p[i] = 0.0f;
}

// flat f32 -> bf16
__global__ void cvt_bf16_kernel(const float* __restrict__ X, unsigned short* __restrict__ O,
                                long long n) {
    long long i = (long long)blockIdx.x * blockDim.x + threadIdx.x;
    long long stride = (long long)gridDim.x * blockDim.x;
    for (; i < n; i += stride) O[i] = f2bf(X[i]);
}

// W [K][Ncol] f32 -> WT [Ncol][K] bf16
__global__ void cvt_transpose(const float* __restrict__ W, unsigned short* __restrict__ WT,
                              int K, int Ncol) {
    int i = blockIdx.x * blockDim.x + threadIdx.x;
    if (i >= K * Ncol) return;
    int k = i / Ncol, n = i % Ncol;
    WT[n * K + k] = f2bf(W[i]);
}

// out_bf16[i] = cvt( relu?( X[i] + b[i % ld] ) )
__global__ void bias_act_cvt_kernel(const float* __restrict__ X, const float* __restrict__ b,
                                    unsigned short* __restrict__ O, long long n, int ld,
                                    int do_relu) {
    long long i = (long long)blockIdx.x * blockDim.x + threadIdx.x;
    long long stride = (long long)gridDim.x * blockDim.x;
    for (; i < n; i += stride) {
        float v = X[i] + b[i % ld];
        if (do_relu) v = v > 0.0f ? v : 0.0f;
        O[i] = f2bf(v);
    }
}

// C[M][Ncol] = A[M][K] (bf16) x BT[Ncol][K] (bf16) [+ bias]
// One wave32 per 16x32 output tile: one A fragment feeds two v_wmma_f32_16x16x32_bf16.
template <int K>
__global__ void wmma_gemm(const unsigned short* __restrict__ A,
                          const unsigned short* __restrict__ BT,
                          float* __restrict__ C, const float* __restrict__ bias,
                          int M, int Ncol) {
    int wave = (blockIdx.x * blockDim.x + threadIdx.x) >> 5;
    int lane = threadIdx.x & 31;
    int ntn = Ncol >> 5;                       // 32-column tiles
    int ntiles = (M >> 4) * ntn;
    if (wave >= ntiles) return;                // wave-uniform: EXEC all-1s for WMMA
    int tm = wave / ntn, tn = wave % ntn;
    int l15 = lane & 15, half = lane >> 4;

    const unsigned short* ap  = A  + (long long)((tm << 4) + l15) * K;  // A row
    const unsigned short* bp0 = BT + (long long)((tn << 5) + l15) * K;  // B cols [0,16)
    const unsigned short* bp1 = bp0 + (long long)16 * K;                // B cols [16,32)

    v8f acc0 = {}, acc1 = {};
#pragma unroll
    for (int kb = 0; kb < K; kb += 32) {
        // A fragment: lanes 0-15 hold K[0,16), lanes 16-31 hold K[16,32) of row
        union { v16bf v; uint4 q[2]; } af, bf0, bf1;
        af.q[0]  = *(const uint4*)(ap + kb + half * 16);
        af.q[1]  = *(const uint4*)(ap + kb + half * 16 + 8);
        // B fragment: lanes 0-15 hold K[0,16), lanes 16-31 hold K[16,32) of column
        bf0.q[0] = *(const uint4*)(bp0 + kb + half * 16);
        bf0.q[1] = *(const uint4*)(bp0 + kb + half * 16 + 8);
        bf1.q[0] = *(const uint4*)(bp1 + kb + half * 16);
        bf1.q[1] = *(const uint4*)(bp1 + kb + half * 16 + 8);
        acc0 = __builtin_amdgcn_wmma_f32_16x16x32_bf16(false, af.v, false, bf0.v,
                                                       (short)0, acc0, false, false);
        acc1 = __builtin_amdgcn_wmma_f32_16x16x32_bf16(false, af.v, false, bf1.v,
                                                       (short)0, acc1, false, false);
    }

    // C/D layout: VGPR v -> row (v + half*8), col = lane&15
    float* cp = C + (long long)(tm << 4) * Ncol + (tn << 5) + l15;
    float bv0 = bias ? bias[(tn << 5) + l15] : 0.0f;
    float bv1 = bias ? bias[(tn << 5) + 16 + l15] : 0.0f;
#pragma unroll
    for (int v = 0; v < 8; v++) {
        long long r = (long long)(v + (half << 3)) * Ncol;
        cp[r]      = acc0[v] + bv0;
        cp[r + 16] = acc1[v] + bv1;
    }
}

// alpha_src[n,h] = sum_c H[n,h,c]*a_src[h,c]  (and same for dst)
__global__ void alpha_kernel(const float* __restrict__ Hm, const float* __restrict__ a_src,
                             const float* __restrict__ a_dst, float* __restrict__ AS,
                             float* __restrict__ AD, int Nn, int heads, int ch) {
    int i = blockIdx.x * blockDim.x + threadIdx.x;
    if (i >= Nn * heads) return;
    int n = i / heads, h = i % heads;
    const float4* hp = (const float4*)(Hm + (long long)n * heads * ch + h * ch);
    const float4* sp = (const float4*)(a_src + h * ch);
    const float4* dp = (const float4*)(a_dst + h * ch);
    float s = 0.0f, d = 0.0f;
    for (int c = 0; c < (ch >> 2); c++) {
        float4 v = hp[c], av = sp[c], bv = dp[c];
        s += v.x * av.x + v.y * av.y + v.z * av.z + v.w * av.w;
        d += v.x * bv.x + v.y * bv.y + v.z * bv.z + v.w * bv.w;
    }
    AS[i] = s; AD[i] = d;
}

__device__ __forceinline__ void edge_sd(const int* ei, int E0, int e, int& s, int& d) {
    if (e < E0) { s = ei[e]; d = ei[E0 + e]; } else { s = d = e - E0; }
}

__global__ void edge_max_kernel(const int* __restrict__ ei, int E0, int Etot, int heads,
                                const float* __restrict__ AS, const float* __restrict__ AD,
                                unsigned* __restrict__ EMAX) {
    int i = blockIdx.x * blockDim.x + threadIdx.x;
    if (i >= Etot * heads) return;
    int e = i / heads, h = i % heads;
    int s, d; edge_sd(ei, E0, e, s, d);
    float v = lrelu(AS[s * heads + h] + AD[d * heads + h]);
    atomicMax(&EMAX[d * heads + h], fkey(v));
}

__global__ void edge_expsum_kernel(const int* __restrict__ ei, int E0, int Etot, int heads,
                                   const float* __restrict__ AS, const float* __restrict__ AD,
                                   const unsigned* __restrict__ EMAX, float* __restrict__ WB,
                                   float* __restrict__ DEN) {
    int i = blockIdx.x * blockDim.x + threadIdx.x;
    if (i >= Etot * heads) return;
    int e = i / heads, h = i % heads;
    int s, d; edge_sd(ei, E0, e, s, d);
    float v = lrelu(AS[s * heads + h] + AD[d * heads + h]);
    float w = __expf(v - funkey(EMAX[d * heads + h]));
    WB[i] = w;
    atomicAdd(&DEN[d * heads + h], w);
}

// one wave32 per (edge, head): out[dst, h, :] += (w/denom) * H[src, h, :]
__global__ void edge_scatter_kernel(const int* __restrict__ ei, int E0, int Etot, int heads,
                                    int ch, const float* __restrict__ WB,
                                    const float* __restrict__ DEN,
                                    const float* __restrict__ Hm, float* __restrict__ OUT) {
    int g = (blockIdx.x * blockDim.x + threadIdx.x) >> 5;
    int lane = threadIdx.x & 31;
    if (g >= Etot * heads) return;
    int e = g / heads, h = g % heads;
    int s, d; edge_sd(ei, E0, e, s, d);
    float coef = WB[g] / (DEN[d * heads + h] + GAT_EPS);
    const float* hs = Hm + (long long)s * heads * ch + h * ch;
    float*       od = OUT + (long long)d * heads * ch + h * ch;
    for (int c = lane; c < ch; c += 32)
        atomicAdd(&od[c], coef * hs[c]);
}

extern "C" void kernel_launch(void* const* d_in, const int* in_sizes, int n_in,
                              void* d_out, int out_size, void* d_ws, size_t ws_size,
                              hipStream_t stream) {
    const float* x   = (const float*)d_in[0];
    const int*   ei  = (const int*)d_in[1];
    const float* W1  = (const float*)d_in[2];
    const float* as1 = (const float*)d_in[3];
    const float* ad1 = (const float*)d_in[4];
    const float* b1  = (const float*)d_in[5];
    const float* W2  = (const float*)d_in[6];
    const float* as2 = (const float*)d_in[7];
    const float* ad2 = (const float*)d_in[8];
    const float* b2  = (const float*)d_in[9];
    const float* Wl  = (const float*)d_in[10];
    const float* bl  = (const float*)d_in[11];
    float* out = (float*)d_out;

    const int IN_CH = 128, HID = 64, HEADS = 4, OUT_CH = 64, HC1 = HEADS * HID; // 256
    const int Nn   = in_sizes[0] / IN_CH;  // 50000
    const int E0   = in_sizes[1] / 2;      // 800000
    const int Etot = E0 + Nn;              // + self loops

    char* ws = (char*)d_ws;
    size_t off = 0;
    auto alloc = [&](size_t bytes) -> char* {
        char* p = ws + off;
        off = (off + bytes + 255) & ~(size_t)255;
        return p;
    };
    unsigned short* W1T = (unsigned short*)alloc((size_t)IN_CH * HC1 * 2);
    unsigned short* W2T = (unsigned short*)alloc((size_t)HC1 * HID * 2);
    unsigned short* WlT = (unsigned short*)alloc((size_t)HID * OUT_CH * 2);
    float*    AS1   = (float*)alloc((size_t)Nn * HEADS * 4);
    float*    AD1   = (float*)alloc((size_t)Nn * HEADS * 4);
    unsigned* EMAX1 = (unsigned*)alloc((size_t)Nn * HEADS * 4);
    float*    DEN1  = (float*)alloc((size_t)Nn * HEADS * 4);
    float*    AS2   = (float*)alloc((size_t)Nn * 4);
    float*    AD2   = (float*)alloc((size_t)Nn * 4);
    unsigned* EMAX2 = (unsigned*)alloc((size_t)Nn * 4);
    float*    DEN2  = (float*)alloc((size_t)Nn * 4);
    float*    WB    = (float*)alloc((size_t)Etot * HEADS * 4);  // edge weights (reused L2)
    float*    H1    = (float*)alloc((size_t)Nn * HC1 * 4);      // conv1 pre-agg features
    float*    X2    = (float*)alloc((size_t)Nn * HC1 * 4);      // conv1 aggregated output
    float*    O2    = (float*)alloc((size_t)Nn * HID * 4);      // conv2 aggregated output
    // one bf16 staging region, time-shared: xb (GEMM1 A) -> X2b (GEMM2 A) -> O2b (GEMM3 A)
    unsigned short* BFB = (unsigned short*)alloc((size_t)Nn * HC1 * 2);
    float* H2 = H1;                                             // reuse (H1 dead after scatter1)

    const int B = 256;
    auto cdiv = [](long long a, long long b) { return (int)((a + b - 1) / b); };

    // weight transforms (f32 -> transposed bf16)
    cvt_transpose<<<cdiv((long long)IN_CH * HC1, B), B, 0, stream>>>(W1, W1T, IN_CH, HC1);
    cvt_transpose<<<cdiv((long long)HC1 * HID, B), B, 0, stream>>>(W2, W2T, HC1, HID);
    cvt_transpose<<<cdiv((long long)HID * OUT_CH, B), B, 0, stream>>>(Wl, WlT, HID, OUT_CH);

    // zero accumulators + segment buffers
    zero_kernel<<<2048, B, 0, stream>>>(X2, (long long)Nn * HC1);
    zero_kernel<<<1024, B, 0, stream>>>(O2, (long long)Nn * HID);
    zero_kernel<<<256, B, 0, stream>>>((float*)EMAX1, (long long)Nn * HEADS);
    zero_kernel<<<256, B, 0, stream>>>(DEN1, (long long)Nn * HEADS);
    zero_kernel<<<256, B, 0, stream>>>((float*)EMAX2, (long long)Nn);
    zero_kernel<<<256, B, 0, stream>>>(DEN2, (long long)Nn);

    // ---- layer 1: conv1 (heads=4, ch=64) ----
    cvt_bf16_kernel<<<2048, B, 0, stream>>>(x, BFB, (long long)Nn * IN_CH);     // xb
    long long tiles1 = (long long)(Nn / 16) * (HC1 / 32);
    wmma_gemm<IN_CH><<<cdiv(tiles1 * 32, B), B, 0, stream>>>(BFB, W1T, H1, nullptr, Nn, HC1);
    alpha_kernel<<<cdiv((long long)Nn * HEADS, B), B, 0, stream>>>(H1, as1, ad1, AS1, AD1, Nn, HEADS, HID);
    edge_max_kernel<<<cdiv((long long)Etot * HEADS, B), B, 0, stream>>>(ei, E0, Etot, HEADS, AS1, AD1, EMAX1);
    edge_expsum_kernel<<<cdiv((long long)Etot * HEADS, B), B, 0, stream>>>(ei, E0, Etot, HEADS, AS1, AD1, EMAX1, WB, DEN1);
    edge_scatter_kernel<<<cdiv((long long)Etot * HEADS * 32, B), B, 0, stream>>>(ei, E0, Etot, HEADS, HID, WB, DEN1, H1, X2);
    // X2b = bf16(relu(X2 + b1))   (xb dead -> reuse staging region)
    bias_act_cvt_kernel<<<2048, B, 0, stream>>>(X2, b1, BFB, (long long)Nn * HC1, HC1, 1);

    // ---- layer 2: conv2 (heads=1, ch=64) ----
    long long tiles2 = (long long)(Nn / 16) * (HID / 32);
    wmma_gemm<HC1><<<cdiv(tiles2 * 32, B), B, 0, stream>>>(BFB, W2T, H2, nullptr, Nn, HID);
    alpha_kernel<<<cdiv((long long)Nn, B), B, 0, stream>>>(H2, as2, ad2, AS2, AD2, Nn, 1, HID);
    edge_max_kernel<<<cdiv((long long)Etot, B), B, 0, stream>>>(ei, E0, Etot, 1, AS2, AD2, EMAX2);
    edge_expsum_kernel<<<cdiv((long long)Etot, B), B, 0, stream>>>(ei, E0, Etot, 1, AS2, AD2, EMAX2, WB, DEN2);
    edge_scatter_kernel<<<cdiv((long long)Etot * 32, B), B, 0, stream>>>(ei, E0, Etot, 1, HID, WB, DEN2, H2, O2);
    // O2b = bf16(O2 + b2)   (X2b dead -> reuse staging region)
    bias_act_cvt_kernel<<<1024, B, 0, stream>>>(O2, b2, BFB, (long long)Nn * HID, HID, 0);

    // ---- final linear: out = O2 @ Wl + bl ----
    long long tiles3 = (long long)(Nn / 16) * (OUT_CH / 32);
    wmma_gemm<HID><<<cdiv(tiles3 * 32, B), B, 0, stream>>>(BFB, WlT, out, bl, Nn, OUT_CH);
}
